// CustomAttention_35278861369702
// MI455X (gfx1250) — matrix operand order repaired and checked by reading, not compiled, not run on hardware
//
#include <hip/hip_runtime.h>
#include <hip/hip_bf16.h>
#include <stdint.h>

#define B_   2
#define S_   2048
#define H_   2048
#define NH_  16
#define NKV_ 4
#define HD_  128

typedef __attribute__((ext_vector_type(16))) __bf16 v16bf;
typedef __attribute__((ext_vector_type(8)))  __bf16 v8bf;
typedef __attribute__((ext_vector_type(8)))  float  v8f;
typedef __attribute__((ext_vector_type(4)))  unsigned v4u;
typedef __attribute__((ext_vector_type(8)))  int      v8i;
typedef __attribute__((ext_vector_type(4)))  int      v4i;

static __device__ __forceinline__ __bf16 f2bf(float f) {
  unsigned u = __builtin_bit_cast(unsigned, f);
  unsigned r = u + 0x7FFFu + ((u >> 16) & 1u);   // round-to-nearest-even
  unsigned short h = (unsigned short)(r >> 16);
  return __builtin_bit_cast(__bf16, h);
}

// ---- WMMA fragment loaders (wave32, 16x16x32 bf16 layouts per CDNA5 ISA) ----
// A 16x32 (MxK): lane holds row m=lane&15; element e -> K = half*8 + (e&7) + ((e&8)<<1)
static __device__ __forceinline__ v16bf load_a_frag(const __bf16* base, int stride, int lane) {
  int m = lane & 15, half = lane >> 4;
  const __bf16* p0 = base + (size_t)m * stride + half * 8;
  v8bf lo = *(const v8bf*)p0;
  v8bf hi = *(const v8bf*)(p0 + 16);
  v16bf r;
#pragma unroll
  for (int i = 0; i < 8; ++i) { r[i] = lo[i]; r[i + 8] = hi[i]; }
  return r;
}

// B 32x16 (KxN): lane holds column n=lane&15, K = (lane>>4)*8 + (e&7) + ((e&8)<<1); source is [n][k] row-major
static __device__ __forceinline__ v16bf load_b_frag(const __bf16* base, int stride, int lane) {
  int n = lane & 15, kh = lane >> 4;
  const __bf16* p = base + (size_t)n * stride + kh * 8;
  v8bf lo = *(const v8bf*)p;
  v8bf hi = *(const v8bf*)(p + 16);
  v16bf r;
#pragma unroll
  for (int i = 0; i < 8; ++i) { r[i] = lo[i]; r[i + 8] = hi[i]; }
  return r;
}

static __device__ __forceinline__ v8f wmma_bf16(v16bf a, v16bf b, v8f c) {
  return __builtin_amdgcn_wmma_f32_16x16x32_bf16(false, a, false, b, (short)0, c, false, false);
}

// ---- CDNA5 async memory->LDS copy (ASYNCcnt), 16B per lane ----
static __device__ __forceinline__ void async_ld_b128(const void* g, void* l) {
  unsigned lds = (unsigned)(uintptr_t)l;                 // LDS offset = addr[31:0]
  unsigned long long ga = (unsigned long long)(uintptr_t)g;
  asm volatile("global_load_async_to_lds_b128 %0, %1, off"
               :: "v"(lds), "v"(ga) : "memory");
}
static __device__ __forceinline__ void wait_async0() {
  asm volatile("s_wait_asynccnt 0" ::: "memory");
}

// ---- CDNA5 Tensor Data Mover: DMA a 2D tile (rows x 64B) into padded LDS ----
// tile_dim0 = 32 bf16 (64B) per row; LDS gets 16B pad per row -> 80B (LDA=40) stride.
static __device__ __forceinline__ void tdm_load_2d(const void* gptr, void* lptr,
                                                   int rows, int stride_elems) {
  unsigned long long ga = (unsigned long long)(uintptr_t)gptr;
  unsigned ga_lo = __builtin_amdgcn_readfirstlane((unsigned)ga);
  unsigned ga_hi = __builtin_amdgcn_readfirstlane((unsigned)(ga >> 32));
  unsigned lds   = __builtin_amdgcn_readfirstlane((unsigned)(uintptr_t)lptr);
  unsigned urows = __builtin_amdgcn_readfirstlane((unsigned)rows);
  v4u g0;
  g0[0] = 1u;                                   // count=1 (valid), not restore/gather
  g0[1] = lds;                                  // lds_addr (bytes)
  g0[2] = ga_lo;                                // global_addr[31:0]
  g0[3] = (ga_hi & 0x1FFFFFFu) | 0x80000000u;   // global_addr[56:32] | type=2
  v8i g1;
  g1[0] = (int)((1u << 16)     // data_size = 2B
              | (1u << 20)     // pad_enable
              | (3u << 22)     // pad_interval: 16 DWORDs (one 64B row)
              | (3u << 25));   // pad_amount: 4 DWORDs (16B)
  g1[1] = (int)(32u << 16);            // tensor_dim0 = 32 (low16 @ [31:16])
  g1[2] = (int)(urows << 16);          // tensor_dim0 hi=0 | tensor_dim1 low16
  g1[3] = (int)(32u << 16);            // tensor_dim1 hi=0 | tile_dim0 = 32
  g1[4] = (int)urows;                  // tile_dim1 = rows | tile_dim2 = 0
  g1[5] = (int)(unsigned)stride_elems; // tensor_dim0_stride low32 (data_size units)
  g1[6] = 0;                           // stride0 hi | stride1 lo
  g1[7] = 0;                           // stride1 hi
  v4i z = {};
#if __clang_major__ >= 23
  v8i z8 = {};
  __builtin_amdgcn_tensor_load_to_lds(g0, g1, z, z, z8, 0);
#else
  __builtin_amdgcn_tensor_load_to_lds(g0, g1, z, z, 0);
#endif
}
static __device__ __forceinline__ void wait_tensor0() {
  __builtin_amdgcn_s_wait_tensorcnt(0);
}

// ---- fp32 -> bf16 elementwise ----
__global__ void cvt_f32_bf16(const float* __restrict__ src, __bf16* __restrict__ dst, int n) {
  int i = blockIdx.x * blockDim.x + threadIdx.x;
  if (i < n) dst[i] = f2bf(src[i]);
}

// ---- GEMM: C[M,N] = A[M,K](bf16) @ W[N,K]^T(bf16), fp32 out.
// 8 waves/WG, 128x128 tile, double-buffered TDM (tensor_load_to_lds) staging:
// waves 0-3 DMA 32-row slices of A tile, waves 4-7 DMA the B tile.
#define LDA 40   // 32 K elems + 8 pad (matches TDM pad: 16B per 64B row)
__global__ __launch_bounds__(256) void gemm_wmma(const __bf16* __restrict__ A,
                                                 const __bf16* __restrict__ W,
                                                 float* __restrict__ C,
                                                 int N, int K, int mode, int nheads) {
  __shared__ __align__(16) __bf16 Ab[2][128 * LDA];
  __shared__ __align__(16) __bf16 Bb[2][128 * LDA];
  int tid = threadIdx.x;
  int lane = tid & 31, wid = tid >> 5;
  int wm = wid & 1, wn = wid >> 1;            // wave -> 64x32 output subtile
  int n0 = blockIdx.x * 128, m0 = blockIdx.y * 128;
  const __bf16* Abase = A + (size_t)m0 * K;
  const __bf16* Wbase = W + (size_t)n0 * K;

  auto stage = [&](int buf, int k0) {
    if (wid < 4) {
      int row0 = wid * 32;
      tdm_load_2d(Abase + (size_t)row0 * K + k0, &Ab[buf][row0 * LDA], 32, K);
    } else {
      int row0 = (wid - 4) * 32;
      tdm_load_2d(Wbase + (size_t)row0 * K + k0, &Bb[buf][row0 * LDA], 32, K);
    }
  };

  v8f acc[4][2] = {};
  stage(0, 0);
  wait_tensor0();
  __syncthreads();
  int nsteps = K / 32;
  for (int s = 0; s < nsteps; ++s) {
    int cur = s & 1;
    if (s + 1 < nsteps) stage(cur ^ 1, (s + 1) * 32);
    v16bf af[4], bw[2];
#pragma unroll
    for (int i = 0; i < 4; ++i) af[i] = load_a_frag(&Ab[cur][(wm * 64 + i * 16) * LDA], LDA, lane);
#pragma unroll
    for (int j = 0; j < 2; ++j) bw[j] = load_b_frag(&Bb[cur][(wn * 32 + j * 16) * LDA], LDA, lane);
#pragma unroll
    for (int i = 0; i < 4; ++i)
#pragma unroll
      for (int j = 0; j < 2; ++j) acc[i][j] = wmma_bf16(af[i], bw[j], acc[i][j]);
    wait_tensor0();
    __syncthreads();
  }

  int half = lane >> 4, nl = lane & 15;
#pragma unroll
  for (int i = 0; i < 4; ++i) {
#pragma unroll
    for (int j = 0; j < 2; ++j) {
#pragma unroll
      for (int r = 0; r < 8; ++r) {
        int m = m0 + wm * 64 + i * 16 + r + 8 * half;
        int n = n0 + wn * 32 + j * 16 + nl;
        size_t off;
        if (mode == 0) {
          off = (size_t)m * N + n;
        } else {
          int b = m / S_, s = m % S_;
          int h = n / HD_, d = n % HD_;
          off = (((size_t)b * nheads + h) * S_ + s) * HD_ + d;
        }
        C[off] = acc[i][j][r];
      }
    }
  }
}

// ---- RoPE + fp32->bf16, rows of [*, HD]; thread handles the (d, d+64) pair ----
__global__ void rope_cvt(const float* __restrict__ X, __bf16* __restrict__ Y, int nrows) {
  int idx = blockIdx.x * blockDim.x + threadIdx.x;
  if (idx >= nrows * 64) return;
  int d = idx & 63;
  int row = idx >> 6;
  int s = row & (S_ - 1);
  float inv_freq = __powf(10000.0f, -(float)(2 * d) * (1.0f / 128.0f));
  float ang = (float)s * inv_freq;
  float c, sn;
  __sincosf(ang, &c, &sn);
  size_t base = (size_t)row * HD_;
  float lo = X[base + d], hi = X[base + d + 64];
  Y[base + d]      = f2bf(lo * c - hi * sn);
  Y[base + d + 64] = f2bf(hi * c + lo * sn);
}

// ---- V [B,NKV,S,HD] fp32 -> V^T [B,NKV,HD,S] bf16 ----
__global__ void v_transpose(const float* __restrict__ V, __bf16* __restrict__ Vt, int n) {
  int idx = blockIdx.x * blockDim.x + threadIdx.x;
  if (idx >= n) return;
  int d = idx & (HD_ - 1);
  int t = idx >> 7;
  int s = t & (S_ - 1);
  int bh = t >> 11;
  Vt[((size_t)bh * HD_ + d) * S_ + s] = f2bf(V[idx]);
}

// ---- Flash attention: 8 waves/WG = 128 query rows for one (b,h); K/V tiles
// shared through double-buffered async-to-LDS (ASYNCcnt); causal; GQA. ----
#define LDK 136  // 128 d + 8 pad
#define LDV 40   // 32 k + 8 pad
#define LDP 40
__global__ __launch_bounds__(256) void flash_attn(const __bf16* __restrict__ Qr,
                                                  const __bf16* __restrict__ Kr,
                                                  const __bf16* __restrict__ Vt,
                                                  __bf16* __restrict__ O) {
  __shared__ __align__(16) __bf16 Kl[2][32 * LDK];
  __shared__ __align__(16) __bf16 Vl[2][128 * LDV];
  __shared__ __align__(16) __bf16 Pl[8][16 * LDP];
  int tid = threadIdx.x, lane = tid & 31, wid = tid >> 5;
  const int WGQ = S_ / 128;   // 16 query groups per (b,h)
  int wg = blockIdx.x;
  int wgq = wg % WGQ;
  int bh = wg / WGQ;
  int h = bh % NH_;
  int b = bh / NH_;
  int hkv = h / (NH_ / NKV_);
  int qbase = wgq * 128 + wid * 16;
  const __bf16* Qbase = Qr + (((size_t)(b * NH_ + h)) * S_ + qbase) * HD_;
  const __bf16* Kbase = Kr + ((size_t)(b * NKV_ + hkv)) * S_ * HD_;
  const __bf16* Vbase = Vt + ((size_t)(b * NKV_ + hkv)) * HD_ * S_;

  v16bf qf[4];
#pragma unroll
  for (int c = 0; c < 4; ++c) qf[c] = load_a_frag(Qbase + c * 32, HD_, lane);

  v8f acc[8] = {};
  float rowm[8], rowl[8];
#pragma unroll
  for (int r = 0; r < 8; ++r) { rowm[r] = -1e30f; rowl[r] = 0.0f; }

  int half = lane >> 4, nl = lane & 15;
  int qrow0 = qbase + 8 * half;
  const float scale = 0.08838834764831845f;  // 1/sqrt(128)

  auto stageKV = [&](int buf, int kbase) {
#pragma unroll
    for (int i = 0; i < 2; ++i) {
      int c = tid + i * 256;
      int krow = c >> 4, kc = c & 15;        // K tile: 32 rows x 256B
      async_ld_b128(Kbase + (size_t)(kbase + krow) * HD_ + kc * 8,
                    &Kl[buf][krow * LDK + kc * 8]);
      int vrow = c >> 2, vc = c & 3;         // V^T tile: 128 rows x 64B
      async_ld_b128(Vbase + (size_t)vrow * S_ + kbase + vc * 8,
                    &Vl[buf][vrow * LDV + vc * 8]);
    }
  };

  int nkt = wgq * 4 + 4;                     // key tiles of 32 covering causal region
  stageKV(0, 0);
  wait_async0();
  __syncthreads();

  for (int kt = 0; kt < nkt; ++kt) {
    int cur = kt & 1;
    int kbase = kt * 32;
    if (kt + 1 < nkt) stageKV(cur ^ 1, (kt + 1) * 32);
    if (kbase <= qbase + 15) {               // wave has unmasked work in this tile
      v8f s0 = {}, s1 = {};
#pragma unroll
      for (int c = 0; c < 4; ++c) {
        v16bf kb0 = load_b_frag(&Kl[cur][0 * LDK + c * 32], LDK, lane);
        v16bf kb1 = load_b_frag(&Kl[cur][16 * LDK + c * 32], LDK, lane);
        s0 = wmma_bf16(qf[c], kb0, s0);
        s1 = wmma_bf16(qf[c], kb1, s1);
      }
#pragma unroll
      for (int r = 0; r < 8; ++r) {
        int qrow = qrow0 + r;
        float a  = s0[r] * scale;
        float bb = s1[r] * scale;
        if (kbase + nl > qrow)      a  = -1e30f;
        if (kbase + 16 + nl > qrow) bb = -1e30f;
        float mx = fmaxf(a, bb);
        mx = fmaxf(mx, __shfl_xor(mx, 1, 32));
        mx = fmaxf(mx, __shfl_xor(mx, 2, 32));
        mx = fmaxf(mx, __shfl_xor(mx, 4, 32));
        mx = fmaxf(mx, __shfl_xor(mx, 8, 32));
        float nm = fmaxf(rowm[r], mx);
        float corr = __expf(rowm[r] - nm);
        float p0 = __expf(a - nm);
        float p1 = __expf(bb - nm);
        float rs = p0 + p1;
        rs += __shfl_xor(rs, 1, 32);
        rs += __shfl_xor(rs, 2, 32);
        rs += __shfl_xor(rs, 4, 32);
        rs += __shfl_xor(rs, 8, 32);
        rowl[r] = rowl[r] * corr + rs;
        rowm[r] = nm;
#pragma unroll
        for (int f = 0; f < 8; ++f) acc[f][r] *= corr;
        Pl[wid][(r + 8 * half) * LDP + nl]      = f2bf(p0);
        Pl[wid][(r + 8 * half) * LDP + 16 + nl] = f2bf(p1);
      }
      // same-wave LDS ops are in-order (DScnt): A-frag load sees the stores
      v16bf pf = load_a_frag(&Pl[wid][0], LDP, lane);
#pragma unroll
      for (int f = 0; f < 8; ++f) {
        v16bf vb = load_b_frag(&Vl[cur][(f * 16) * LDV], LDV, lane);
        acc[f] = wmma_bf16(pf, vb, acc[f]);
      }
    }
    wait_async0();
    __syncthreads();
  }

#pragma unroll
  for (int f = 0; f < 8; ++f) {
#pragma unroll
    for (int r = 0; r < 8; ++r) {
      float o = acc[f][r] / rowl[r];
      int srow = qrow0 + r;
      int col = h * HD_ + f * 16 + nl;
      O[((size_t)b * S_ + srow) * (NH_ * HD_) + col] = f2bf(o);
    }
  }
}

extern "C" void kernel_launch(void* const* d_in, const int* in_sizes, int n_in,
                              void* d_out, int out_size, void* d_ws, size_t ws_size,
                              hipStream_t stream) {
  (void)in_sizes; (void)n_in; (void)out_size; (void)ws_size;
  const float* x  = (const float*)d_in[0];
  // d_in[1] = attention_mask: causal, applied analytically in flash_attn
  const float* Wq = (const float*)d_in[2];
  const float* Wk = (const float*)d_in[3];
  const float* Wv = (const float*)d_in[4];
  const float* Wo = (const float*)d_in[5];
  float* out = (float*)d_out;

  char* ws = (char*)d_ws;
  size_t off = 0;
  auto alloc = [&](size_t bytes) -> char* {
    char* p = ws + off;
    off += (bytes + 255) & ~(size_t)255;
    return p;
  };

  const size_t nx  = (size_t)B_ * S_ * H_;         // 8,388,608
  const size_t nwq = (size_t)NH_ * HD_ * H_;       // 4,194,304
  const size_t nwk = (size_t)NKV_ * HD_ * H_;      // 1,048,576
  const size_t nQ  = (size_t)B_ * NH_ * S_ * HD_;  // 8,388,608
  const size_t nK  = (size_t)B_ * NKV_ * S_ * HD_; // 2,097,152

  __bf16* xb  = (__bf16*)alloc(nx  * 2);
  __bf16* Wqb = (__bf16*)alloc(nwq * 2);
  __bf16* Wkb = (__bf16*)alloc(nwk * 2);
  __bf16* Wvb = (__bf16*)alloc(nwk * 2);
  __bf16* Wob = (__bf16*)alloc(nwq * 2);
  float*  Qf  = (float*) alloc(nQ * 4);
  float*  Kf  = (float*) alloc(nK * 4);
  float*  Vf  = (float*) alloc(nK * 4);
  __bf16* Qrb = (__bf16*)alloc(nQ * 2);
  __bf16* Krb = (__bf16*)alloc(nK * 2);
  __bf16* Vtb = (__bf16*)alloc(nK * 2);
  __bf16* Ob  = (__bf16*)alloc(nx * 2);

  // 1) bf16 conversions
  cvt_f32_bf16<<<(int)((nx  + 255) / 256), 256, 0, stream>>>(x,  xb,  (int)nx);
  cvt_f32_bf16<<<(int)((nwq + 255) / 256), 256, 0, stream>>>(Wq, Wqb, (int)nwq);
  cvt_f32_bf16<<<(int)((nwk + 255) / 256), 256, 0, stream>>>(Wk, Wkb, (int)nwk);
  cvt_f32_bf16<<<(int)((nwk + 255) / 256), 256, 0, stream>>>(Wv, Wvb, (int)nwk);
  cvt_f32_bf16<<<(int)((nwq + 255) / 256), 256, 0, stream>>>(Wo, Wob, (int)nwq);

  // 2) QKV projections (WMMA + TDM staging), scatter to [B, heads, S, HD] fp32
  const int M = B_ * S_;  // 4096
  gemm_wmma<<<dim3((NH_ * HD_) / 128, M / 128), 256, 0, stream>>>(xb, Wqb, Qf, NH_ * HD_, H_, 1, NH_);
  gemm_wmma<<<dim3((NKV_ * HD_) / 128, M / 128), 256, 0, stream>>>(xb, Wkb, Kf, NKV_ * HD_, H_, 1, NKV_);
  gemm_wmma<<<dim3((NKV_ * HD_) / 128, M / 128), 256, 0, stream>>>(xb, Wvb, Vf, NKV_ * HD_, H_, 1, NKV_);

  // 3) RoPE + bf16 for Q,K; transpose + bf16 for V
  int qrows = B_ * NH_ * S_;   // 65536
  int krows = B_ * NKV_ * S_;  // 16384
  rope_cvt<<<(qrows * 64 + 255) / 256, 256, 0, stream>>>(Qf, Qrb, qrows);
  rope_cvt<<<(krows * 64 + 255) / 256, 256, 0, stream>>>(Kf, Krb, krows);
  v_transpose<<<(int)((nK + 255) / 256), 256, 0, stream>>>(Vf, Vtb, (int)nK);

  // 4) causal flash attention (WMMA + shared async K/V tiles) -> O bf16 [B*S, NH*HD]
  flash_attn<<<B_ * NH_ * (S_ / 128), 256, 0, stream>>>(Qrb, Krb, Vtb, Ob);

  // 5) output projection (WMMA) -> fp32 d_out [B*S, H]
  gemm_wmma<<<dim3(H_ / 128, M / 128), 256, 0, stream>>>(Ob, Wob, out, H_, H_, 0, 0);
}
